// Classifier_temporal_MTGNN_5729486373633
// MI455X (gfx1250) — matrix operand order repaired and checked by reading, not compiled, not run on hardware
//
#include <hip/hip_runtime.h>

// ---------------- problem constants ----------------
#define BB   4
#define CIN  129
#define NN   1259
#define TT   25
#define T1   23
#define RC   128
#define CC   126
#define SCc  128
#define OUTD 64

#define NT      (NN*TT)        // 31475
#define XBATCH  (CIN*NT)       // 4060275
#define H0BATCH (RC*NT)        // 4028800
#define FCH     (NN*T1)        // 28957
#define FBATCH  (CC*FCH)       // 3648582
#define HOPB    (CC*T1)        // 2898
#define HOPJ    (BB*HOPB)      // 11592
#define MPCH    (NN*T1)        // 28957
#define MPBATCH (RC*MPCH)      // 3706496
#define SKIPB   (SCc*NN)       // 161152
#define LNBLK   14479          // ceil(MPBATCH/256)

#define WN 2                   // 16x16 tiles per wave along N

typedef __attribute__((ext_vector_type(16))) __bf16    v16bf;
typedef __attribute__((ext_vector_type(8)))  __bf16    v8bf;
typedef __attribute__((ext_vector_type(2)))  __bf16    v2bf;
typedef __attribute__((ext_vector_type(2)))  float     v2f;
typedef __attribute__((ext_vector_type(8)))  float     v8f;
typedef __attribute__((ext_vector_type(4)))  unsigned  v4u;
typedef __attribute__((ext_vector_type(4)))  int       v4i;
typedef __attribute__((ext_vector_type(8)))  int       v8i;

__device__ __forceinline__ long koff(int k, int krad, long sKc, long sKd) {
  if (krad == 1) return (long)k * sKc;
  int kc = k / krad;
  return (long)kc * sKc + (long)(k - kc * krad) * sKd;
}

// ---------------- generic WMMA GEMM ----------------
// C[m, j] = act( outScale * sum_k A[m,k]*B(k,j) + xScale*X[m,j] + bias[m] (+ C[m,j] if accum) )
// B(k,j):  addr = colBase(j) + kOff(k);  j -> (d2,d1,d0) via radices (r1,r0);
//          colBase = d0*sB0 + d1*sB1 + d2*sB2;  kOff = (k/krad)*sKc + (k%krad)*sKd
// C / X:   addr = m*sCm + d0*sC0 + d1*sC1 + d2*sC2
struct GemmArgs {
  const float* A;  long lda;
  const float* Bm;
  const float* bias;
  const float* Xadd; float xScale; float outScale;
  float* C;
  int M, K, Ncols;
  int r0, r1;
  long sB0, sB1, sB2;
  long sC0, sC1, sC2, sCm;
  int krad; long sKc, sKd;
  int accum, act, reluB;
};

__global__ __launch_bounds__(256)
void gemm_wmma_bf16(GemmArgs g) {
  // Double-buffered staging: TDM drops the raw f32 16x32 A tile into ldsF,
  // then the block repacks to bf16 in WMMA A-fragment order in ldsA.
  __shared__ __align__(16) float  ldsF[2][512];
  __shared__ __align__(16) __bf16 ldsA[2][512];

  const int tid  = threadIdx.x;
  const int lane = tid & 31;
  const int m0   = blockIdx.y * 16;
  const long colUnit = (long)blockIdx.x * 8 + (tid >> 5);
  const int n0 = (int)(colUnit * (16 * WN));

  const bool waveActive   = (n0 < g.Ncols);                 // wave-uniform
  const bool waveFullCols = (n0 + 16 * WN <= g.Ncols);      // wave-uniform
  const bool blockFullM   = (m0 + 16 <= g.M);               // block-uniform

  long colB[WN], colC[WN];
  bool cOK[WN];
#pragma unroll
  for (int w = 0; w < WN; ++w) {
    int n = n0 + w * 16 + (lane & 15);
    cOK[w] = (n < g.Ncols);
    int nn = cOK[w] ? n : 0;
    int d0 = nn % g.r0; int rest = nn / g.r0;
    int d1 = rest % g.r1; int d2 = rest / g.r1;
    colB[w] = (long)d0 * g.sB0 + (long)d1 * g.sB1 + (long)d2 * g.sB2;
    colC[w] = (long)d0 * g.sC0 + (long)d1 * g.sC1 + (long)d2 * g.sC2;
  }

  const int kbB = (lane < 16) ? 0 : 16;   // B k = 16*(lane/16) + i, same contiguous k order as A

  // repack pair indices (constant per thread): elements 2*tid, 2*tid+1
  const int reM  = (tid * 2) >> 5;
  const int reKl = (tid * 2) & 31;        // even
  const int reDst = (reM + (((reKl >> 4) & 1) << 4)) * 16 +
                    (reKl & 15);

  v8f acc[WN] = {};
  int p = 0;

  for (int k0 = 0; k0 < g.K; k0 += 32) {
    const bool kFull = (k0 + 32 <= g.K);

    // ---- TDM: async DMA of the 16x32 f32 A tile into LDS (wave 0 issues) ----
    // D# per ISA 8.3/8.4: 2D tile, data_size=4B, tile_dim 32x16, stride=lda,
    // tensor_dim = remaining extents (hardware zero-fills OOB reads).
    if ((tid >> 5) == 0) {
      unsigned long ga = (unsigned long)(const void*)(g.A + (long)m0 * g.lda + k0);
      unsigned ldsoff = (unsigned)(unsigned long)(const void*)&ldsF[p][0];
      unsigned remK = (unsigned)(g.K - k0);
      unsigned remM = (unsigned)(g.M - m0);
      unsigned stride = (unsigned)g.lda;
      v4u g0;
      g0[0] = 1u;                                               // count=1 (valid D#)
      g0[1] = ldsoff;                                           // lds_addr
      g0[2] = (unsigned)(ga & 0xFFFFFFFFu);                     // global_addr lo
      g0[3] = (unsigned)((ga >> 32) & 0x01FFFFFFu) | (2u << 30);// addr hi | type=2
      v8i g1;
      g1[0] = (int)(2u << 16);                                  // data_size=4B
      g1[1] = (int)((remK & 0xFFFFu) << 16);                    // tensor_dim0 lo16
      g1[2] = (int)((remK >> 16) | ((remM & 0xFFFFu) << 16));   // dim0 hi | dim1 lo
      g1[3] = (int)((remM >> 16) | (32u << 16));                // dim1 hi | tile_dim0=32
      g1[4] = 16;                                               // tile_dim1=16, tile_dim2=0
      g1[5] = (int)stride;                                      // tensor_dim0_stride lo32
      g1[6] = 0;                                                // stride hi | dim1_stride lo
      g1[7] = 0;
      v4i z4 = {};
#if defined(__clang_major__) && (__clang_major__ >= 23)
      v8i z8 = {};
      __builtin_amdgcn_tensor_load_to_lds(g0, g1, z4, z4, z8, 0);
#else
      __builtin_amdgcn_tensor_load_to_lds(g0, g1, z4, z4, 0);
#endif
      __builtin_amdgcn_s_wait_tensorcnt(0);
    }
    __syncthreads();

    // ---- repack f32 -> bf16 pairs into A-fragment order ----
    // (one ds_load_b64 + one v_cvt_pk_bf16_f32 + one ds_store_b32 per thread)
    {
      v2f vv = *(const v2f*)&ldsF[p][tid * 2];
      v2bf pk;
      pk[0] = (__bf16)vv[0];
      pk[1] = (__bf16)vv[1];
      *(v2bf*)&ldsA[p][reDst] = pk;
    }
    __syncthreads();

    if (waveActive) {
      const __bf16* as = &ldsA[p][lane * 16];
      v8bf lo = *(const v8bf*)(as);            // ds_load_b128
      v8bf hi = *(const v8bf*)(as + 8);        // ds_load_b128
      v16bf af = __builtin_shufflevector(lo, hi,
          0, 1, 2, 3, 4, 5, 6, 7, 8, 9, 10, 11, 12, 13, 14, 15);

      if (waveFullCols && kFull) {
        // -------- fast path: no per-lane predicates --------
#pragma unroll
        for (int w = 0; w < WN; ++w) {
          v16bf bfr;
#pragma unroll
          for (int i = 0; i < 16; ++i) {
            int k = k0 + kbB + i;
            float v = g.Bm[colB[w] + koff(k, g.krad, g.sKc, g.sKd)];
            if (g.reluB) v = fmaxf(v, 0.0f);
            bfr[i] = (__bf16)v;
          }
          if (k0 + 32 < g.K)   // prefetch next K-step of this B stream
            __builtin_prefetch(
                (const void*)&g.Bm[colB[w] + koff(k0 + 32 + kbB, g.krad, g.sKc, g.sKd)],
                0, 3);
          acc[w] = __builtin_amdgcn_wmma_f32_16x16x32_bf16(
              false, af, false, bfr, (short)0, acc[w], false, false);
        }
      } else {
        // -------- edge path: guarded --------
#pragma unroll
        for (int w = 0; w < WN; ++w) {
          v16bf bfr;
#pragma unroll
          for (int i = 0; i < 16; ++i) {
            int k = k0 + kbB + i;
            float v = (cOK[w] && (k < g.K))
                          ? g.Bm[colB[w] + koff(k, g.krad, g.sKc, g.sKd)]
                          : 0.0f;
            if (g.reluB) v = fmaxf(v, 0.0f);
            bfr[i] = (__bf16)v;
          }
          acc[w] = __builtin_amdgcn_wmma_f32_16x16x32_bf16(
              false, af, false, bfr, (short)0, acc[w], false, false);
        }
      }
    }
    p ^= 1;
  }

  if (!waveActive) return;

  const int mSt = m0 + ((lane < 16) ? 0 : 8); // C/D: VGPR r -> M=r / r+8 (ISA 7.12.2)
  if (waveFullCols && blockFullM) {
#pragma unroll
    for (int w = 0; w < WN; ++w) {
#pragma unroll
      for (int r = 0; r < 8; ++r) {
        int m = mSt + r;
        long addr = (long)m * g.sCm + colC[w];
        float v = g.outScale * acc[w][r];
        if (g.Xadd)  v += g.xScale * g.Xadd[addr];
        if (g.bias)  v += g.bias[m];
        if (g.accum) v += g.C[addr];
        if (g.act)   v = fmaxf(v, 0.0f);
        g.C[addr] = v;
      }
    }
  } else {
#pragma unroll
    for (int w = 0; w < WN; ++w) {
      if (!cOK[w]) continue;
#pragma unroll
      for (int r = 0; r < 8; ++r) {
        int m = mSt + r;
        if (m >= g.M) continue;
        long addr = (long)m * g.sCm + colC[w];
        float v = g.outScale * acc[w][r];
        if (g.Xadd)  v += g.xScale * g.Xadd[addr];
        if (g.bias)  v += g.bias[m];
        if (g.accum) v += g.C[addr];
        if (g.act)   v = fmaxf(v, 0.0f);
        g.C[addr] = v;
      }
    }
  }
}

static inline void run_gemm(const GemmArgs& g, hipStream_t s) {
  int mt = (g.M + 15) / 16;
  long cu = ((long)g.Ncols + (16 * WN) - 1) / (16 * WN);
  dim3 grid((unsigned)((cu + 7) / 8), (unsigned)mt, 1);
  hipLaunchKernelGGL(gemm_wmma_bf16, grid, dim3(256), 0, s, g);
}

// ---------------- adjacency normalization ----------------
__global__ __launch_bounds__(256)
void norm_adj_kernel(const float* __restrict__ adj,
                     float* __restrict__ a1, float* __restrict__ a2) {
  int row  = blockIdx.x * 8 + (threadIdx.x >> 5);
  int lane = threadIdx.x & 31;
  if (row >= NN) return;
  float s1 = 0.f, s2 = 0.f;
  for (int c = lane; c < NN; c += 32) {
    float d = (c == row) ? 1.0f : 0.0f;
    s1 += adj[(long)row * NN + c] + d;
    s2 += adj[(long)c * NN + row] + d;
  }
#pragma unroll
  for (int o = 16; o; o >>= 1) {
    s1 += __shfl_xor(s1, o, 32);
    s2 += __shfl_xor(s2, o, 32);
  }
  float r1 = 1.0f / s1, r2 = 1.0f / s2;
  for (int c = lane; c < NN; c += 32) {
    float d = (c == row) ? 1.0f : 0.0f;
    a1[(long)row * NN + c] = (adj[(long)row * NN + c] + d) * r1;
    a2[(long)row * NN + c] = (adj[(long)c * NN + row] + d) * r2;
  }
}

// ---------------- gate fusion + transpose to hop layout ----------------
// hop0T[n, b, c, t] = tanh(filt[b,c,n,t]) * sigmoid(gate[b,c,n,t])
__global__ __launch_bounds__(256)
void combine_kernel(const float* __restrict__ filt, const float* __restrict__ gate,
                    float* __restrict__ hop0T) {
  long idx = (long)blockIdx.x * 256 + threadIdx.x;
  if (idx >= (long)BB * FBATCH) return;
  int t = (int)(idx % T1); long r = idx / T1;
  int n = (int)(r % NN);   r /= NN;
  int c = (int)(r % CC);   int b = (int)(r / CC);
  float f = tanhf(filt[idx]);
  float gv = 1.0f / (1.0f + __expf(-gate[idx]));
  hop0T[(long)n * HOPJ + (long)b * HOPB + c * T1 + t] = f * gv;
}

// ---------------- residual add + LayerNorm (3 deterministic stages) ----------------
__global__ __launch_bounds__(256)
void res_ln1(float* __restrict__ mp, const float* __restrict__ h0,
             float* __restrict__ pS, float* __restrict__ pQ) {
  int b = blockIdx.z;
  long base = (long)b * MPBATCH;
  long idx = (long)blockIdx.x * 256 + threadIdx.x;
  float s = 0.f, q = 0.f;
  if (idx < MPBATCH) {
    int t = (int)(idx % T1); long r = idx / T1;
    int n = (int)(r % NN);   int c = (int)(r / NN);
    float res = h0[(long)b * H0BATCH + (long)c * NT + n * TT + (t + 2)];
    float v = mp[base + idx] + res;
    mp[base + idx] = v;
    s = v; q = v * v;
  }
  __shared__ float sS[256], sQ[256];
  sS[threadIdx.x] = s; sQ[threadIdx.x] = q;
  __syncthreads();
  for (int o = 128; o; o >>= 1) {
    if (threadIdx.x < (unsigned)o) {
      sS[threadIdx.x] += sS[threadIdx.x + o];
      sQ[threadIdx.x] += sQ[threadIdx.x + o];
    }
    __syncthreads();
  }
  if (threadIdx.x == 0) {
    pS[(long)b * LNBLK + blockIdx.x] = sS[0];
    pQ[(long)b * LNBLK + blockIdx.x] = sQ[0];
  }
}

__global__ __launch_bounds__(256)
void res_ln2(const float* __restrict__ pS, const float* __restrict__ pQ,
             float* __restrict__ stats) {
  int b = blockIdx.x;
  float s = 0.f, q = 0.f;
  for (int i = threadIdx.x; i < LNBLK; i += 256) {
    s += pS[(long)b * LNBLK + i];
    q += pQ[(long)b * LNBLK + i];
  }
  __shared__ float sS[256], sQ[256];
  sS[threadIdx.x] = s; sQ[threadIdx.x] = q;
  __syncthreads();
  for (int o = 128; o; o >>= 1) {
    if (threadIdx.x < (unsigned)o) {
      sS[threadIdx.x] += sS[threadIdx.x + o];
      sQ[threadIdx.x] += sQ[threadIdx.x + o];
    }
    __syncthreads();
  }
  if (threadIdx.x == 0) {
    float inv = 1.0f / (float)MPBATCH;
    float mean = sS[0] * inv;
    float var = sQ[0] * inv - mean * mean;
    stats[b * 2 + 0] = mean;
    stats[b * 2 + 1] = rsqrtf(var + 1e-5f);
  }
}

__global__ __launch_bounds__(256)
void res_ln3(float* __restrict__ mp, const float* __restrict__ stats) {
  int b = blockIdx.z;
  long idx = (long)blockIdx.x * 256 + threadIdx.x;
  if (idx >= MPBATCH) return;
  long a = (long)b * MPBATCH + idx;
  mp[a] = (mp[a] - stats[b * 2]) * stats[b * 2 + 1];
}

// ---------------- final linear head + sigmoid ----------------
__global__ __launch_bounds__(256)
void head_kernel(const float* __restrict__ o2, const float* __restrict__ w_head,
                 const float* __restrict__ b_head, float* __restrict__ out) {
  int i = blockIdx.x * 256 + threadIdx.x;  // over B*NN
  if (i >= BB * NN) return;
  int b = i / NN, n = i % NN;
  float s = b_head[0];
#pragma unroll 8
  for (int d = 0; d < OUTD; ++d)
    s += o2[(long)b * (OUTD * NN) + (long)d * NN + n] * w_head[d];
  out[i] = 1.0f / (1.0f + __expf(-s));
}

// ---------------- host orchestration ----------------
extern "C" void kernel_launch(void* const* d_in, const int* in_sizes, int n_in,
                              void* d_out, int out_size, void* d_ws, size_t ws_size,
                              hipStream_t stream) {
  (void)in_sizes; (void)n_in; (void)out_size; (void)ws_size;
  const float* x       = (const float*)d_in[0];
  const float* adj     = (const float*)d_in[1];
  const float* w_start = (const float*)d_in[2];
  const float* b_start = (const float*)d_in[3];
  const float* w_filt  = (const float*)d_in[4];
  const float* b_filt  = (const float*)d_in[5];
  const float* w_gate  = (const float*)d_in[6];
  const float* b_gate  = (const float*)d_in[7];
  const float* w_skip0 = (const float*)d_in[8];
  const float* b_skip0 = (const float*)d_in[9];
  const float* w_skip1 = (const float*)d_in[10];
  const float* b_skip1 = (const float*)d_in[11];
  const float* w_mp1   = (const float*)d_in[12];
  const float* b_mp1   = (const float*)d_in[13];
  const float* w_mp2   = (const float*)d_in[14];
  const float* b_mp2   = (const float*)d_in[15];
  const float* w_skipE = (const float*)d_in[16];
  const float* b_skipE = (const float*)d_in[17];
  const float* w_end1  = (const float*)d_in[18];
  const float* b_end1  = (const float*)d_in[19];
  const float* w_end2  = (const float*)d_in[20];
  const float* b_end2  = (const float*)d_in[21];
  const float* w_head  = (const float*)d_in[22];
  const float* b_head  = (const float*)d_in[23];
  float* out = (float*)d_out;

  // workspace carve-up
  float* w = (float*)d_ws;
  float* h0    = w; w += (long)BB * H0BATCH;      // [b,128,n,25]
  float* a1    = w; w += (long)NN * NN;
  float* a2    = w; w += (long)NN * NN;
  float* bufP  = w; w += (long)BB * FBATCH;       // filt raw / hop ping
  float* bufQ  = w; w += (long)BB * FBATCH;       // gate raw / hop pong
  float* hop0  = w; w += (long)NN * HOPJ;         // gated output, transposed [n,b,c,t]
  float* mp    = w; w += (long)BB * MPBATCH;      // mixprop accum -> h1 (in place)
  float* skip  = w; w += (long)BB * SKIPB;        // [b,128,n]
  float* o1    = w; w += (long)BB * SKIPB;
  float* o2    = w; w += (long)BB * OUTD * NN;
  float* pS    = w; w += (long)BB * LNBLK;
  float* pQ    = w; w += (long)BB * LNBLK;
  float* stats = w; w += 8;

  // 1) adjacency normalization (both directions)
  hipLaunchKernelGGL(norm_adj_kernel, dim3((NN + 7) / 8), dim3(256), 0, stream, adj, a1, a2);

  GemmArgs g{};

  // 2) start conv1x1: h0 = W_start @ x + b   (M=128, K=129, N=B*NT)
  g = GemmArgs{};
  g.A = w_start; g.lda = CIN; g.Bm = x; g.bias = b_start; g.outScale = 1.f; g.C = h0;
  g.M = RC; g.K = CIN; g.Ncols = BB * NT;
  g.r0 = NT; g.r1 = BB;
  g.sB0 = 1; g.sB1 = XBATCH; g.sB2 = 0;
  g.sC0 = 1; g.sC1 = H0BATCH; g.sC2 = 0; g.sCm = NT;
  g.krad = 1; g.sKc = NT; g.sKd = 0;
  run_gemm(g, stream);

  // 3) skip0 = tconv(x, w_skip0)  (M=128, K=129*25, N=B*NN) -> skip (write)
  g = GemmArgs{};
  g.A = w_skip0; g.lda = CIN * TT; g.Bm = x; g.bias = b_skip0; g.outScale = 1.f; g.C = skip;
  g.M = SCc; g.K = CIN * TT; g.Ncols = BB * NN;
  g.r0 = NN; g.r1 = BB;
  g.sB0 = TT; g.sB1 = XBATCH; g.sB2 = 0;
  g.sC0 = 1; g.sC1 = SKIPB; g.sC2 = 0; g.sCm = NN;
  g.krad = TT; g.sKc = NT; g.sKd = 1;
  run_gemm(g, stream);

  // 4/5) filter and gate temporal convs (M=126, K=128*3, N=B*NN*T1)
  for (int which = 0; which < 2; ++which) {
    g = GemmArgs{};
    g.A = which ? w_gate : w_filt; g.lda = RC * 3; g.Bm = h0;
    g.bias = which ? b_gate : b_filt; g.outScale = 1.f; g.C = which ? bufQ : bufP;
    g.M = CC; g.K = RC * 3; g.Ncols = BB * NN * T1;
    g.r0 = T1; g.r1 = NN;
    g.sB0 = 1; g.sB1 = TT; g.sB2 = H0BATCH;
    g.sC0 = 1; g.sC1 = T1; g.sC2 = FBATCH; g.sCm = FCH;
    g.krad = 3; g.sKc = NT; g.sKd = 1;
    run_gemm(g, stream);
  }

  // 6) gated fusion -> hop0 (transposed layout)
  {
    long tot = (long)BB * FBATCH;
    hipLaunchKernelGGL(combine_kernel, dim3((unsigned)((tot + 255) / 256)), dim3(256), 0,
                       stream, bufP, bufQ, hop0);
  }

  // 7) skip += tconv(fg, w_skip1)  (M=128, K=126*23, N=B*NN)
  g = GemmArgs{};
  g.A = w_skip1; g.lda = CC * T1; g.Bm = hop0; g.bias = b_skip1; g.outScale = 1.f; g.C = skip;
  g.M = SCc; g.K = CC * T1; g.Ncols = BB * NN;
  g.r0 = NN; g.r1 = BB;
  g.sB0 = HOPJ; g.sB1 = HOPB; g.sB2 = 0;
  g.sC0 = 1; g.sC1 = SKIPB; g.sC2 = 0; g.sCm = NN;
  g.krad = 1; g.sKc = 1; g.sKd = 0;
  g.accum = 1;
  run_gemm(g, stream);

  // mixprop helpers
  auto mp_conv = [&](const float* hop, const float* wmpBase, int gchunk,
                     const float* bias, int accum) {
    GemmArgs a{};
    a.A = wmpBase + gchunk * CC; a.lda = 4 * CC; a.Bm = hop; a.bias = bias;
    a.outScale = 1.f; a.C = mp;
    a.M = RC; a.K = CC; a.Ncols = BB * NN * T1;
    a.r0 = T1; a.r1 = NN;
    a.sB0 = 1; a.sB1 = HOPJ; a.sB2 = HOPB;
    a.sC0 = 1; a.sC1 = T1; a.sC2 = MPBATCH; a.sCm = MPCH;
    a.krad = 1; a.sKc = T1; a.sKd = 0;
    a.accum = accum;
    run_gemm(a, stream);
  };
  auto spmm = [&](const float* anorm, const float* hin, float* hout) {
    GemmArgs a{};
    a.A = anorm; a.lda = NN; a.Bm = hin;
    a.Xadd = hop0; a.xScale = 0.5f; a.outScale = 0.5f; a.C = hout;
    a.M = NN; a.K = NN; a.Ncols = HOPJ;
    a.r0 = HOPJ; a.r1 = 1;
    a.sB0 = 1; a.sB1 = 0; a.sB2 = 0;
    a.sC0 = 1; a.sC1 = 0; a.sC2 = 0; a.sCm = HOPJ;
    a.krad = 1; a.sKc = HOPJ; a.sKd = 0;
    run_gemm(a, stream);
  };

  // 8) mixprop #1 (A = a1): hops 0..3, conv-accumulate into mp
  mp_conv(hop0, w_mp1, 0, b_mp1, /*accum=*/0);
  spmm(a1, hop0, bufP);  mp_conv(bufP, w_mp1, 1, nullptr, 1);
  spmm(a1, bufP, bufQ);  mp_conv(bufQ, w_mp1, 2, nullptr, 1);
  spmm(a1, bufQ, bufP);  mp_conv(bufP, w_mp1, 3, nullptr, 1);

  // 9) mixprop #2 (A = a2)
  mp_conv(hop0, w_mp2, 0, b_mp2, /*accum=*/1);
  spmm(a2, hop0, bufP);  mp_conv(bufP, w_mp2, 1, nullptr, 1);
  spmm(a2, bufP, bufQ);  mp_conv(bufQ, w_mp2, 2, nullptr, 1);
  spmm(a2, bufQ, bufP);  mp_conv(bufP, w_mp2, 3, nullptr, 1);

  // 10) residual add + LayerNorm (in place on mp)
  hipLaunchKernelGGL(res_ln1, dim3(LNBLK, 1, BB), dim3(256), 0, stream, mp, h0, pS, pQ);
  hipLaunchKernelGGL(res_ln2, dim3(BB), dim3(256), 0, stream, pS, pQ, stats);
  hipLaunchKernelGGL(res_ln3, dim3(LNBLK, 1, BB), dim3(256), 0, stream, mp, stats);

  // 11) skip += tconv(h1, w_skipE)  (M=128, K=128*23, N=B*NN)
  g = GemmArgs{};
  g.A = w_skipE; g.lda = RC * T1; g.Bm = mp; g.bias = b_skipE; g.outScale = 1.f; g.C = skip;
  g.M = SCc; g.K = RC * T1; g.Ncols = BB * NN;
  g.r0 = NN; g.r1 = BB;
  g.sB0 = T1; g.sB1 = MPBATCH; g.sB2 = 0;
  g.sC0 = 1; g.sC1 = SKIPB; g.sC2 = 0; g.sCm = NN;
  g.krad = T1; g.sKc = MPCH; g.sKd = 1;
  g.accum = 1;
  run_gemm(g, stream);

  // 12) o1 = relu(W_end1 @ relu(skip) + b)   (M=128, K=128, N=B*NN)
  g = GemmArgs{};
  g.A = w_end1; g.lda = SCc; g.Bm = skip; g.bias = b_end1; g.outScale = 1.f; g.C = o1;
  g.M = 128; g.K = SCc; g.Ncols = BB * NN;
  g.r0 = NN; g.r1 = BB;
  g.sB0 = 1; g.sB1 = SKIPB; g.sB2 = 0;
  g.sC0 = 1; g.sC1 = SKIPB; g.sC2 = 0; g.sCm = NN;
  g.krad = 1; g.sKc = NN; g.sKd = 0;
  g.reluB = 1; g.act = 1;
  run_gemm(g, stream);

  // 13) o2 = W_end2 @ o1 + b   (M=64, K=128, N=B*NN)
  g = GemmArgs{};
  g.A = w_end2; g.lda = 128; g.Bm = o1; g.bias = b_end2; g.outScale = 1.f; g.C = o2;
  g.M = OUTD; g.K = 128; g.Ncols = BB * NN;
  g.r0 = NN; g.r1 = BB;
  g.sB0 = 1; g.sB1 = SKIPB; g.sB2 = 0;
  g.sC0 = 1; g.sC1 = (long)OUTD * NN; g.sC2 = 0; g.sCm = NN;
  g.krad = 1; g.sKc = NN; g.sKd = 0;
  run_gemm(g, stream);

  // 14) sigmoid head -> d_out [B, N]
  hipLaunchKernelGGL(head_kernel, dim3((BB * NN + 255) / 256), dim3(256), 0, stream,
                     o2, w_head, b_head, out);
}
